// MultiHead_Global_Attention_59794534695005
// MI455X (gfx1250) — compile-verified, matrix operand order checked by reading, not run on hardware
//
#include <hip/hip_runtime.h>
#include <hip/hip_bf16.h>
#include <math.h>

typedef _Float16 half_t;
typedef __attribute__((ext_vector_type(16))) _Float16 v16h;
typedef __attribute__((ext_vector_type(8)))  _Float16 v8h;
typedef __attribute__((ext_vector_type(8)))  float    v8f;
typedef __attribute__((ext_vector_type(4)))  int      v4i;

#define B_   4
#define N_   2048
#define C_   256
#define H_   8
#define ATT_ 512
#define DH_  64
#define BH_  (B_ * H_)
#define BN_  (B_ * N_)

// ---- CDNA5 async global->LDS staging (guarded; fallback = sync copy) --------
#if defined(__gfx1250__) && __has_builtin(__builtin_amdgcn_global_load_async_to_lds_b128)
#define HAVE_ASYNC 1
#else
#define HAVE_ASYNC 0
#endif

#if HAVE_ASYNC
typedef __attribute__((address_space(1))) v4i* as1_v4i;
typedef __attribute__((address_space(3))) v4i* as3_v4i;
#endif

__device__ __forceinline__ void cp16(const half_t* g, half_t* l) {
#if HAVE_ASYNC
  __builtin_amdgcn_global_load_async_to_lds_b128((as1_v4i)g, (as3_v4i)l, 0, 0);
#else
  *(uint4*)l = *(const uint4*)g;
#endif
}

template <int Nw>
__device__ __forceinline__ void wait_async() {
#if HAVE_ASYNC
#if __has_builtin(__builtin_amdgcn_s_wait_asynccnt)
  __builtin_amdgcn_s_wait_asynccnt(Nw);
#else
  if constexpr (Nw == 0) asm volatile("s_wait_asynccnt 0x0" ::: "memory");
  else if constexpr (Nw == 1) asm volatile("s_wait_asynccnt 0x1" ::: "memory");
  else if constexpr (Nw == 4) asm volatile("s_wait_asynccnt 0x4" ::: "memory");
  else asm volatile("s_wait_asynccnt 0x8" ::: "memory");
#endif
#endif
}

// ---- WMMA helpers -----------------------------------------------------------
// A-matrix 16x32 f16 layout (CDNA5 ISA 7.12.2): lane<16 holds K={0..7,16..23},
// lane>=16 holds K={8..15,24..31} for its row M = lane%16. B mirrors with
// N = lane%16 as the column.
__device__ __forceinline__ int kpat(int i, int hi) {
  return (i < 8 ? i : i + 8) + hi * 8;
}
__device__ __forceinline__ v8f wmma_f16(v16h a, v16h b, v8f c) {
  return __builtin_amdgcn_wmma_f32_16x16x32_f16(false, a, false, b, (short)0, c,
                                                false, false);
}
__device__ __forceinline__ v16h frag_row_h(const half_t* base, int hi) {
  v16h f;
#pragma unroll
  for (int i = 0; i < 16; ++i) f[i] = base[kpat(i, hi)];
  return f;
}

// ---- Kernel 0a: convert all weights to f16 (all six are 131072 elements) ---
// W1 stays in its native [DH][K] layout: that IS the transposed (d-major)
// layout the P@W1^T B-fragments want.
__global__ void k_convert_w(const float* __restrict__ W0, const float* __restrict__ W1,
                            const float* __restrict__ Wq, const float* __restrict__ Wk,
                            const float* __restrict__ Wv, const float* __restrict__ Wp,
                            half_t* __restrict__ W0h, half_t* __restrict__ W1h,
                            half_t* __restrict__ Wqh, half_t* __restrict__ Wkh,
                            half_t* __restrict__ Wvh, half_t* __restrict__ Wph) {
  int idx = blockIdx.x * 256 + threadIdx.x;
  if (idx >= N_ * DH_) return;
  W0h[idx] = (half_t)W0[idx];
  W1h[idx] = (half_t)W1[idx];
  Wqh[idx] = (half_t)Wq[idx];
  Wkh[idx] = (half_t)Wk[idx];
  Wvh[idx] = (half_t)Wv[idx];
  Wph[idx] = (half_t)Wp[idx];
}

// ---- Kernel 0b: convert x to f16 -------------------------------------------
__global__ void k_convert_x(const float* __restrict__ x, half_t* __restrict__ xh) {
  int idx = blockIdx.x * 256 + threadIdx.x;
  if (idx < BN_ * C_) xh[idx] = (half_t)x[idx];
}

// ---- Kernel 1: Q/K/U projections (f16 outputs) ------------------------------
// z=0: Q [bh][n][d] ; z=1: K [bh][n][d]
// z=2: U^T [bh][d][n] = V[n][d] + W1[d][n]   (out1+out3 fusion, d-major so the
//      k_main B-fragments are contiguous; epilogue stores are 8 consecutive n)
__global__ void __launch_bounds__(128)
k_qkvu(const half_t* __restrict__ xh,
       const half_t* __restrict__ Wqh, const half_t* __restrict__ Wkh,
       const half_t* __restrict__ Wvh,
       const float* __restrict__ bq, const float* __restrict__ bk,
       const float* __restrict__ bv, const float* __restrict__ W1,
       half_t* __restrict__ Qh, half_t* __restrict__ Kh, half_t* __restrict__ Uth) {
  __shared__ half_t XL[2][64 * 32];
  __shared__ half_t WL[2][64 * 32];
  const int tid = threadIdx.x;
  const int lane = tid & 31, wave = tid >> 5;
  const int r = lane & 15, hi = lane >> 4;
  const int rowB = blockIdx.x * 64;               // block rows in [0, B*N)
  const int row0 = rowB + wave * 16;
  const int o0 = blockIdx.y * 64;                 // output cols in [0, ATT)
  const int z = blockIdx.z;
  const half_t* Wsel = (z == 0) ? Wqh : (z == 1) ? Wkh : Wvh;
  const float*  bias = (z == 0) ? bq : (z == 1) ? bk : bv;

  auto issue = [&](int buf, int kk) {
#pragma unroll
    for (int j = 0; j < 2; ++j) {
      int c = tid + j * 128;                      // 256 16B chunks per tile
      int row = c >> 2, cc = c & 3;
      cp16(&xh[(size_t)(rowB + row) * C_ + kk + cc * 8], &XL[buf][row * 32 + cc * 8]);
      cp16(&Wsel[(size_t)(o0 + row) * C_ + kk + cc * 8], &WL[buf][row * 32 + cc * 8]);
    }
  };

  v8f acc[4] = {};
  issue(0, 0);
  for (int it = 0; it < C_ / 32; ++it) {
    const int cur = it & 1;
    if (it + 1 < C_ / 32) { issue(cur ^ 1, (it + 1) * 32); wait_async<4>(); }
    else wait_async<0>();
    __syncthreads();
    v16h a = frag_row_h(&XL[cur][(wave * 16 + r) * 32], hi);
#pragma unroll
    for (int s = 0; s < 4; ++s) {
      v16h bf = frag_row_h(&WL[cur][(s * 16 + r) * 32], hi);
      acc[s] = wmma_f16(a, bf, acc[s]);
    }
    __syncthreads();
  }
#pragma unroll
  for (int s = 0; s < 4; ++s) {
    const int o = o0 + s * 16 + r;                // D-tile: col = lane%16
    const int h = o / DH_, d = o % DH_;
    const float bo = bias[o];
#pragma unroll
    for (int v = 0; v < 8; ++v) {
      const int rr = row0 + v + 8 * hi;           // D-tile: row = v + 8*(lane/16)
      const int bb = rr / N_, n = rr % N_;
      const int bh = bb * H_ + h;
      float val = acc[s][v] + bo;
      if (z == 2) {
        val += W1[(size_t)d * N_ + n];            // U = V + W1^T  (stored d-major)
        Uth[((size_t)bh * DH_ + d) * N_ + n] = (half_t)val;
      } else {
        half_t* outp = (z == 0) ? Qh : Kh;
        outp[((size_t)bh * N_ + n) * DH_ + d] = (half_t)val;
      }
    }
  }
}

// ---- Kernel 2: column-softmax stats: off[m] = colmax + log(colsumexp) -------
// z=0: S1 = Q W0^T ; z=1: S2 = Q K^T.  Streams S once, keeps only O(N) stats.
__global__ void __launch_bounds__(128)
k_stats(const half_t* __restrict__ Qh, const half_t* __restrict__ Kh,
        const half_t* __restrict__ W0h,
        float* __restrict__ off1, float* __restrict__ off2) {
  __shared__ half_t QL[2][16 * 64];
  const int tid = threadIdx.x;
  const int lane = tid & 31, wave = tid >> 5;
  const int r = lane & 15, hi = lane >> 4;
  const int m0 = blockIdx.x * 64 + wave * 16;     // this wave's 16 columns
  const int bh = blockIdx.y;
  const int z = blockIdx.z;
  const half_t* Bsrc = (z == 0) ? W0h : (Kh + (size_t)bh * N_ * DH_);
  float* offp = (z == 0) ? off1 : off2;
  const half_t* Qb = Qh + (size_t)bh * N_ * DH_;

  // B-frags fixed for the whole n-loop (columns fixed, k-dim = DH = 64)
  v16h b0 = frag_row_h(&Bsrc[(size_t)(m0 + r) * DH_ + 0], hi);
  v16h b1 = frag_row_h(&Bsrc[(size_t)(m0 + r) * DH_ + 32], hi);

  auto issue = [&](int buf, int n0) {             // one contiguous 2KB block
    cp16(&Qb[(size_t)n0 * DH_ + tid * 8], &QL[buf][tid * 8]);
  };

  float runmax = -3.0e38f, runsum = 0.f;          // per-lane: one column, half rows
  issue(0, 0);
  for (int it = 0; it < N_ / 16; ++it) {
    const int cur = it & 1;
    if (it + 1 < N_ / 16) { issue(cur ^ 1, (it + 1) * 16); wait_async<1>(); }
    else wait_async<0>();
    __syncthreads();
    v16h a0 = frag_row_h(&QL[cur][r * DH_ + 0], hi);
    v16h a1 = frag_row_h(&QL[cur][r * DH_ + 32], hi);
    v8f c = {};
    c = wmma_f16(a0, b0, c);
    c = wmma_f16(a1, b1, c);
    float tmax = c[0];
#pragma unroll
    for (int v = 1; v < 8; ++v) tmax = fmaxf(tmax, c[v]);
    float nm = fmaxf(runmax, tmax);
    runsum *= __expf(runmax - nm);
#pragma unroll
    for (int v = 0; v < 8; ++v) runsum += __expf(c[v] - nm);
    runmax = nm;
    __syncthreads();
  }
  // combine the two row-halves of each column (lane L <-> L+16), wave32 shuffle
  float om = __shfl_xor(runmax, 16, 32);
  float os = __shfl_xor(runsum, 16, 32);
  float nm = fmaxf(runmax, om);
  float total = runsum * __expf(runmax - nm) + os * __expf(om - nm);
  if (hi == 0) offp[(size_t)bh * N_ + m0 + r] = nm + __logf(fmaxf(total, 1e-30f));
}

// ---- Kernel 3: fused recompute + softmax + P@U + P@W1t ----------------------
// Per wave: 16 rows n; loop m over 2048 in chunks of 32. All four operand tiles
// are async-staged and double-buffered; U^T/W1 are d-major so every fragment
// read is a contiguous ds_load_b128 pair (no strided gathers).
__global__ void __launch_bounds__(128)
k_main(const half_t* __restrict__ Qh, const half_t* __restrict__ Kh,
       const half_t* __restrict__ Uth, const half_t* __restrict__ W0h,
       const half_t* __restrict__ W1h,
       const float* __restrict__ off1, const float* __restrict__ off2,
       half_t* __restrict__ hout) {
  __shared__ half_t TL[2][4][32 * DH_];  // W0 | K (m-major), Ut | W1 (d-major)
  __shared__ half_t Pb[4][2][16 * 32];   // per-wave: P1,P2 staging (D->A transpose)
  const int tid = threadIdx.x;
  const int lane = tid & 31, wave = tid >> 5;
  const int r = lane & 15, hi = lane >> 4;
  const int row0 = blockIdx.x * 64 + wave * 16;
  const int bh = blockIdx.y;
  const int bb = bh / H_, h = bh % H_;

  const half_t* Qb  = Qh  + (size_t)bh * N_ * DH_;
  const half_t* Kb  = Kh  + (size_t)bh * N_ * DH_;
  const half_t* Utb = Uth + (size_t)bh * DH_ * N_;
  const float* o1p = off1 + (size_t)bh * N_;
  const float* o2p = off2 + (size_t)bh * N_;

  v16h qa0 = frag_row_h(&Qb[(size_t)(row0 + r) * DH_ + 0], hi);
  v16h qa1 = frag_row_h(&Qb[(size_t)(row0 + r) * DH_ + 32], hi);

  v8f acc13[4] = {}, acc2[4] = {};
  float rowsum = 0.f;  // lane<16: sum_m P1[row r]; lane>=16: sum_m P2[row r]
  half_t* P1 = Pb[wave][0];
  half_t* P2 = Pb[wave][1];

  auto issue = [&](int buf, int m0) {    // 8 async b128 per thread
    // m-major tiles (contiguous 4KB blocks): W0, K rows m0..m0+31
#pragma unroll
    for (int c2 = 0; c2 < 2; ++c2) {
      cp16(W0h + (size_t)m0 * DH_ + tid * 8 + c2 * 1024, &TL[buf][0][tid * 8 + c2 * 1024]);
      cp16(Kb  + (size_t)m0 * DH_ + tid * 8 + c2 * 1024, &TL[buf][1][tid * 8 + c2 * 1024]);
    }
    // d-major tiles: 64 rows x 64B at row stride N_: Ut, W1 cols m0..m0+31
#pragma unroll
    for (int j = 0; j < 2; ++j) {
      int c = tid + j * 128;
      int row = c >> 2, cc = c & 3;
      cp16(&Utb[(size_t)row * N_ + m0 + cc * 8], &TL[buf][2][row * 32 + cc * 8]);
      cp16(&W1h[(size_t)row * N_ + m0 + cc * 8], &TL[buf][3][row * 32 + cc * 8]);
    }
  };

  constexpr int ITERS = N_ / 32;
  issue(0, 0);
  for (int it = 0; it < ITERS; ++it) {
    const int m0 = it * 32;
    const int cur = it & 1;
    if (it + 1 < ITERS) { issue(cur ^ 1, m0 + 32); wait_async<8>(); }
    else wait_async<0>();
    __syncthreads();
    const half_t* TW0 = TL[cur][0];
    const half_t* TK  = TL[cur][1];
    const half_t* TUt = TL[cur][2];
    const half_t* TW1 = TL[cur][3];

#pragma unroll
    for (int ms = 0; ms < 2; ++ms) {
      const int ml = ms * 16 + r;        // local m row / D-tile column
      {  // S1 = Q W0^T tile -> P1
        v16h b0 = frag_row_h(&TW0[ml * DH_ + 0], hi);
        v16h b1 = frag_row_h(&TW0[ml * DH_ + 32], hi);
        v8f c = {};
        c = wmma_f16(qa0, b0, c);
        c = wmma_f16(qa1, b1, c);
        const float o1 = o1p[m0 + ml];
#pragma unroll
        for (int v = 0; v < 8; ++v)
          P1[(v + 8 * hi) * 32 + ml] = (half_t)__expf(c[v] - o1);
      }
      {  // S2 = Q K^T tile -> P2
        v16h b0 = frag_row_h(&TK[ml * DH_ + 0], hi);
        v16h b1 = frag_row_h(&TK[ml * DH_ + 32], hi);
        v8f c = {};
        c = wmma_f16(qa0, b0, c);
        c = wmma_f16(qa1, b1, c);
        const float o2 = o2p[m0 + ml];
#pragma unroll
        for (int v = 0; v < 8; ++v)
          P2[(v + 8 * hi) * 32 + ml] = (half_t)__expf(c[v] - o2);
      }
    }
    // running L1 row sums, vectorized b128 LDS reads (DS in-order within wave)
    {
      const v8h* Prow = (const v8h*)((hi ? P2 : P1) + r * 32);
      float s = 0.f;
#pragma unroll
      for (int j = 0; j < 4; ++j) {
        v8h p = Prow[j];
#pragma unroll
        for (int e = 0; e < 8; ++e) s += (float)p[e];
      }
      rowsum += s;
    }
    // A-frags of P (transposed through LDS), then accumulate P1@U + P2@W1t
    v16h pa1 = frag_row_h(&P1[r * 32], hi);
    v16h pa2 = frag_row_h(&P2[r * 32], hi);
#pragma unroll
    for (int s = 0; s < 4; ++s) {
      v16h ub = frag_row_h(&TUt[(s * 16 + r) * 32], hi);   // B[k=m][col=d], contiguous
      acc13[s] = wmma_f16(pa1, ub, acc13[s]);
      v16h wb = frag_row_h(&TW1[(s * 16 + r) * 32], hi);
      acc2[s] = wmma_f16(pa2, wb, acc2[s]);
    }
    __syncthreads();
  }

#pragma unroll
  for (int s = 0; s < 4; ++s) {
#pragma unroll
    for (int v = 0; v < 8; ++v) {
      const int rowIdx = v + 8 * hi;
      const float rs1 = __shfl(rowsum, rowIdx, 32);
      const float rs2 = __shfl(rowsum, rowIdx + 16, 32);
      const int n = row0 + rowIdx;
      const int d = s * 16 + r;
      float val = acc13[s][v] / (1e-9f + rs1) + acc2[s][v] / (1e-9f + rs2);
      hout[((size_t)bb * N_ + n) * ATT_ + h * DH_ + d] = (half_t)val;
    }
  }
}

// ---- Kernel 4: output projection + residual + ReLU --------------------------
__global__ void __launch_bounds__(128)
k_final(const half_t* __restrict__ hout, const half_t* __restrict__ Wph,
        const float* __restrict__ bp, const float* __restrict__ x,
        float* __restrict__ out) {
  __shared__ half_t AL[2][64 * 32];
  __shared__ half_t BL[2][64 * 32];
  const int tid = threadIdx.x;
  const int lane = tid & 31, wave = tid >> 5;
  const int r = lane & 15, hi = lane >> 4;
  const int rowB = blockIdx.x * 64;
  const int row0 = rowB + wave * 16;
  const int o0 = blockIdx.y * 64;

  auto issue = [&](int buf, int kk) {
#pragma unroll
    for (int j = 0; j < 2; ++j) {
      int c = tid + j * 128;
      int row = c >> 2, cc = c & 3;
      cp16(&hout[(size_t)(rowB + row) * ATT_ + kk + cc * 8], &AL[buf][row * 32 + cc * 8]);
      cp16(&Wph[(size_t)(o0 + row) * ATT_ + kk + cc * 8], &BL[buf][row * 32 + cc * 8]);
    }
  };

  v8f acc[4] = {};
  issue(0, 0);
  for (int it = 0; it < ATT_ / 32; ++it) {
    const int cur = it & 1;
    if (it + 1 < ATT_ / 32) { issue(cur ^ 1, (it + 1) * 32); wait_async<4>(); }
    else wait_async<0>();
    __syncthreads();
    v16h a = frag_row_h(&AL[cur][(wave * 16 + r) * 32], hi);
#pragma unroll
    for (int s = 0; s < 4; ++s) {
      v16h bf = frag_row_h(&BL[cur][(s * 16 + r) * 32], hi);
      acc[s] = wmma_f16(a, bf, acc[s]);
    }
    __syncthreads();
  }
#pragma unroll
  for (int s = 0; s < 4; ++s) {
    const int o = o0 + s * 16 + r;
    const float bo = bp[o];
#pragma unroll
    for (int v = 0; v < 8; ++v) {
      const int rr = row0 + v + 8 * hi;
      float val = acc[s][v] + bo + x[(size_t)rr * C_ + o];
      out[(size_t)rr * C_ + o] = fmaxf(val, 0.f);
    }
  }
}

// ---- Host launch ------------------------------------------------------------
extern "C" void kernel_launch(void* const* d_in, const int* in_sizes, int n_in,
                              void* d_out, int out_size, void* d_ws, size_t ws_size,
                              hipStream_t stream) {
  const float* x  = (const float*)d_in[0];
  const float* Wq = (const float*)d_in[1];
  const float* bq = (const float*)d_in[2];
  const float* Wk = (const float*)d_in[3];
  const float* bk = (const float*)d_in[4];
  const float* Wv = (const float*)d_in[5];
  const float* bv = (const float*)d_in[6];
  const float* W0 = (const float*)d_in[7];
  const float* W1 = (const float*)d_in[8];
  const float* Wp = (const float*)d_in[9];
  const float* bp = (const float*)d_in[10];
  float* out = (float*)d_out;

  char* ws = (char*)d_ws;
  size_t off = 0;
  auto carve = [&](size_t bytes) -> void* {
    char* p = ws + off;
    off += (bytes + 255) & ~(size_t)255;
    return (void*)p;
  };
  half_t* Qh   = (half_t*)carve((size_t)BH_ * N_ * DH_ * 2);  // 8 MB
  half_t* Kh   = (half_t*)carve((size_t)BH_ * N_ * DH_ * 2);  // 8 MB
  half_t* Uth  = (half_t*)carve((size_t)BH_ * N_ * DH_ * 2);  // 8 MB (d-major U^T)
  half_t* W0h  = (half_t*)carve((size_t)N_ * DH_ * 2);        // 256 KB
  half_t* W1h  = (half_t*)carve((size_t)N_ * DH_ * 2);        // 256 KB (d-major)
  half_t* Wqh  = (half_t*)carve((size_t)ATT_ * C_ * 2);       // 256 KB
  half_t* Wkh  = (half_t*)carve((size_t)ATT_ * C_ * 2);       // 256 KB
  half_t* Wvh  = (half_t*)carve((size_t)ATT_ * C_ * 2);       // 256 KB
  half_t* Wph  = (half_t*)carve((size_t)C_ * ATT_ * 2);       // 256 KB
  float*  off1 = (float*)carve((size_t)BH_ * N_ * 4);         // 256 KB
  float*  off2 = (float*)carve((size_t)BH_ * N_ * 4);         // 256 KB
  half_t* xh   = (half_t*)carve((size_t)BN_ * C_ * 2);        // 4 MB
  half_t* hout = (half_t*)carve((size_t)BN_ * ATT_ * 2);      // 8 MB

  k_convert_w<<<dim3((N_ * DH_ + 255) / 256), dim3(256), 0, stream>>>(
      W0, W1, Wq, Wk, Wv, Wp, W0h, W1h, Wqh, Wkh, Wvh, Wph);
  k_convert_x<<<dim3((BN_ * C_ + 255) / 256), dim3(256), 0, stream>>>(x, xh);
  k_qkvu<<<dim3(BN_ / 64, ATT_ / 64, 3), dim3(128), 0, stream>>>(
      xh, Wqh, Wkh, Wvh, bq, bk, bv, W1, Qh, Kh, Uth);
  k_stats<<<dim3(N_ / 64, BH_, 2), dim3(128), 0, stream>>>(Qh, Kh, W0h, off1, off2);
  k_main<<<dim3(N_ / 64, BH_), dim3(128), 0, stream>>>(
      Qh, Kh, Uth, W0h, W1h, off1, off2, hout);
  k_final<<<dim3(BN_ / 64, C_ / 64), dim3(128), 0, stream>>>(hout, Wph, bp, x, out);
}
